// PVCNN2Base_61581241090587
// MI455X (gfx1250) — matrix operand order repaired and checked by reading, not compiled, not run on hardware
//
#include <hip/hip_runtime.h>
#include <math.h>

typedef __attribute__((ext_vector_type(16))) _Float16 v16h;
typedef __attribute__((ext_vector_type(8)))  float    v8f;
typedef __attribute__((ext_vector_type(4)))  float    v4f;

#define EMBED_DIM 64
#define HALF_DIM  32
#define TPTS      64   // points handled per block in the streaming kernel

// ---------------------------------------------------------------------------
// Fragment builders for V_WMMA_F32_16X16X32_F16 per CDNA5 ISA VGPR layouts.
// A (16x32 f16, MxK): lane m=lane&15, kh=lane>>4.
//   element e<8  -> K = 8*kh + e        (VGPR0..3)
//   element e>=8 -> K = 8*kh + e + 8    (VGPR4..7)
// B (32x16 f16, KxN): lane n=lane&15, kh=lane>>4; element e -> same K as A.
// C/D (16x16 f32): acc[v] holds row m = v + 8*kh, col n = lane&15.
// ---------------------------------------------------------------------------
__device__ __forceinline__ v16h load_a_frag(const float* A, int ldA, int k0, int lane) {
  const int m  = lane & 15;
  const int kh = lane >> 4;
  v16h a;
#pragma unroll
  for (int e = 0; e < 16; ++e) {
    const int k = (e < 8) ? (8 * kh + e) : (8 * kh + e + 8);
    a[e] = (_Float16)A[m * ldA + k0 + k];
  }
  return a;
}

// W is row-major [Nout][64]; GEMM uses B[k][n] = W[n0+n][k] (x @ W^T).
__device__ __forceinline__ v16h load_b_frag(const float* W, int n0, int k0, int lane) {
  const int n  = lane & 15;
  const int kh = lane >> 4;
  v16h b;
#pragma unroll
  for (int e = 0; e < 16; ++e) {
    b[e] = (_Float16)W[(n0 + n) * EMBED_DIM + k0 + ((e < 8) ? (8 * kh + e) : (8 * kh + e + 8))];
  }
  return b;
}

// ---------------------------------------------------------------------------
// Setup kernel: one wave (32 lanes). Computes
//   emb[ 0..31] = te  = embed_mlp(sin_emb(t),  ef params)   (row 0 of GEMM A)
//   emb[32..63] = me0 = embed_mlp(sin_emb(0),  em params)   (row 0)
//   emb[64..95] = me1 = embed_mlp(sin_emb(1),  em params)   (row 1)
// Both MLPs are done as 16x16x32 f16 WMMA GEMMs with f32 accumulation.
// ---------------------------------------------------------------------------
__global__ void __launch_bounds__(32)
setup_embed_kernel(const float* __restrict__ t,
                   const float* __restrict__ ef_w1, const float* __restrict__ ef_b1,
                   const float* __restrict__ ef_w2, const float* __restrict__ ef_b2,
                   const float* __restrict__ em_w1, const float* __restrict__ em_b1,
                   const float* __restrict__ em_w2, const float* __restrict__ em_b2,
                   float* __restrict__ emb) {
  __shared__ float A[2][16 * EMBED_DIM];  // [0]=ef input rows, [1]=em input rows
  __shared__ float H[16 * EMBED_DIM];     // hidden activations
  const int lane = threadIdx.x;  // 0..31, one full wave32

  for (int i = lane; i < 16 * EMBED_DIM; i += 32) { A[0][i] = 0.0f; A[1][i] = 0.0f; }
  __syncthreads();

  // sinusoidal embeddings: half=32, freqs[i] = exp(-i * ln(10000)/31)
  {
    const float ls = logf(10000.0f) / (float)(HALF_DIM - 1);
    const float fr = __expf(-(float)lane * ls);
    const float tv = t[0];
    A[0][lane]            = __sinf(tv * fr);       // row 0: t embedding
    A[0][HALF_DIM + lane] = __cosf(tv * fr);
    A[1][lane]            = 0.0f;                  // row 0: mask=0 -> sin(0)=0
    A[1][HALF_DIM + lane] = 1.0f;                  //                  cos(0)=1
    A[1][EMBED_DIM + lane]            = __sinf(fr); // row 1: mask=1
    A[1][EMBED_DIM + HALF_DIM + lane] = __cosf(fr);
  }
  __syncthreads();

  const float* W1[2] = {ef_w1, em_w1};
  const float* B1[2] = {ef_b1, em_b1};
  const float* W2[2] = {ef_w2, em_w2};
  const float* B2[2] = {ef_b2, em_b2};

  for (int s = 0; s < 2; ++s) {
    // layer 1: H = leaky_relu(A(16x64) @ W1^T + b1), N=64 -> 4 tiles, K=64 -> 2 steps
#pragma unroll
    for (int nt = 0; nt < 4; ++nt) {
      v8f acc = {};
#pragma unroll
      for (int ks = 0; ks < 2; ++ks) {
        v16h a = load_a_frag(&A[s][0], EMBED_DIM, ks * 32, lane);
        v16h b = load_b_frag(W1[s], nt * 16, ks * 32, lane);
        acc = __builtin_amdgcn_wmma_f32_16x16x32_f16(false, a, false, b,
                                                     (short)0, acc, false, false);
      }
      const int n  = (lane & 15) + nt * 16;
      const int kh = lane >> 4;
      const float bias = B1[s][n];
#pragma unroll
      for (int v = 0; v < 8; ++v) {
        const float x = acc[v] + bias;
        H[(v + 8 * kh) * EMBED_DIM + n] = (x < 0.0f) ? 0.1f * x : x;  // LeakyReLU(0.1)
      }
    }
    __syncthreads();

    // layer 2: out = H @ W2^T + b2, N=32 -> 2 tiles, K=64 -> 2 steps
#pragma unroll
    for (int nt = 0; nt < 2; ++nt) {
      v8f acc = {};
#pragma unroll
      for (int ks = 0; ks < 2; ++ks) {
        v16h a = load_a_frag(H, EMBED_DIM, ks * 32, lane);
        v16h b = load_b_frag(W2[s], nt * 16, ks * 32, lane);
        acc = __builtin_amdgcn_wmma_f32_16x16x32_f16(false, a, false, b,
                                                     (short)0, acc, false, false);
      }
      const int n  = (lane & 15) + nt * 16;
      const int kh = lane >> 4;
      const float bias = B2[s][n];
      if (kh == 0) {                 // lanes 0..15 hold rows 0 (acc[0]) and 1 (acc[1])
        if (s == 0) {
          emb[n] = acc[0] + bias;            // te
        } else {
          emb[32 + n] = acc[0] + bias;       // me0
          emb[64 + n] = acc[1] + bias;       // me1
        }
      }
    }
    __syncthreads();
  }
}

// ---------------------------------------------------------------------------
// Streaming kernel (bandwidth-bound): each block owns 64 points.
//   out[c*N + p]        = pdf[p][c]                 c in [0,64)   (LDS transpose)
//   out[(64+j)*N + p]   = te[j]                     j in [0,32)
//   out[(96+j)*N + p]   = mask[p] ? me1[j] : me0[j]
// Loads/stores are float4, stores are non-temporal (written once, never re-read).
// ---------------------------------------------------------------------------
__global__ void __launch_bounds__(256)
scatter_kernel(const float* __restrict__ pdf, const int* __restrict__ mask,
               const float* __restrict__ emb, float* __restrict__ out,
               long long n) {
  __shared__ float tile[TPTS][EMBED_DIM + 1];   // +1 pad -> conflict-free transpose
  __shared__ float s_emb[96];
  __shared__ int   s_mask[TPTS];

  const int tid = threadIdx.x;
  const long long p0 = (long long)blockIdx.x * TPTS;
  const long long N  = n;
  const bool vec_ok = ((N & 3) == 0);

  if (tid < 96) s_emb[tid] = emb[tid];
  if (tid < TPTS) {
    const long long p = p0 + tid;
    s_mask[tid] = (p < N) ? mask[p] : 0;
  }

  // load pdf tile: 64 pts x 64 ch = 1024 float4 groups, fully coalesced
#pragma unroll
  for (int j = 0; j < 4; ++j) {
    const int g  = tid + j * 256;     // 0..1023
    const int pr = g >> 4;            // point row 0..63
    const int c  = (g & 15) << 2;     // channel 0,4,...,60
    const long long p = p0 + pr;
    v4f val = {};
    if (p < N) val = __builtin_nontemporal_load((const v4f*)(pdf + p * EMBED_DIM + c));
    tile[pr][c + 0] = val.x;
    tile[pr][c + 1] = val.y;
    tile[pr][c + 2] = val.z;
    tile[pr][c + 3] = val.w;
  }
  __syncthreads();

  // channels 0..63 : transposed pdf
#pragma unroll
  for (int j = 0; j < 4; ++j) {
    const int g = tid + j * 256;
    const int c = g >> 4;             // 0..63
    const int p = (g & 15) << 2;      // 0,4,...,60
    const long long pp = p0 + p;
    v4f val;
    val.x = tile[p + 0][c];
    val.y = tile[p + 1][c];
    val.z = tile[p + 2][c];
    val.w = tile[p + 3][c];
    float* dst = out + (long long)c * N + pp;
    if (vec_ok && (pp + 3 < N)) {
      __builtin_nontemporal_store(val, (v4f*)dst);
    } else {
      if (pp + 0 < N) dst[0] = val.x;
      if (pp + 1 < N) dst[1] = val.y;
      if (pp + 2 < N) dst[2] = val.z;
      if (pp + 3 < N) dst[3] = val.w;
    }
  }

  // channels 64..95 : broadcast te
#pragma unroll
  for (int j = 0; j < 2; ++j) {
    const int g  = tid + j * 256;     // 0..511
    const int jj = g >> 4;            // 0..31
    const int p  = (g & 15) << 2;
    const long long pp = p0 + p;
    const float tv = s_emb[jj];
    v4f val = {tv, tv, tv, tv};
    float* dst = out + (long long)(64 + jj) * N + pp;
    if (vec_ok && (pp + 3 < N)) {
      __builtin_nontemporal_store(val, (v4f*)dst);
    } else {
      if (pp + 0 < N) dst[0] = tv;
      if (pp + 1 < N) dst[1] = tv;
      if (pp + 2 < N) dst[2] = tv;
      if (pp + 3 < N) dst[3] = tv;
    }
  }

  // channels 96..127 : per-point mask-selected embedding
#pragma unroll
  for (int j = 0; j < 2; ++j) {
    const int g  = tid + j * 256;
    const int jj = g >> 4;            // 0..31
    const int p  = (g & 15) << 2;
    const long long pp = p0 + p;
    const float m0v = s_emb[32 + jj];
    const float m1v = s_emb[64 + jj];
    v4f val;
    val.x = s_mask[p + 0] ? m1v : m0v;
    val.y = s_mask[p + 1] ? m1v : m0v;
    val.z = s_mask[p + 2] ? m1v : m0v;
    val.w = s_mask[p + 3] ? m1v : m0v;
    float* dst = out + (long long)(96 + jj) * N + pp;
    if (vec_ok && (pp + 3 < N)) {
      __builtin_nontemporal_store(val, (v4f*)dst);
    } else {
      if (pp + 0 < N) dst[0] = val.x;
      if (pp + 1 < N) dst[1] = val.y;
      if (pp + 2 < N) dst[2] = val.z;
      if (pp + 3 < N) dst[3] = val.w;
    }
  }
}

// ---------------------------------------------------------------------------
// Host launcher. Input order (setup_inputs dict): 0=inputs (unused), 1=t,
// 2=mask, 3=pdf, 4..7=ef params, 8..11=em params.
// d_ws[0..95] holds {te, me0, me1}.
// ---------------------------------------------------------------------------
extern "C" void kernel_launch(void* const* d_in, const int* in_sizes, int n_in,
                              void* d_out, int out_size, void* d_ws, size_t ws_size,
                              hipStream_t stream) {
  const float* t     = (const float*)d_in[1];
  const int*   mask  = (const int*)d_in[2];
  const float* pdf   = (const float*)d_in[3];
  const float* ef_w1 = (const float*)d_in[4];
  const float* ef_b1 = (const float*)d_in[5];
  const float* ef_w2 = (const float*)d_in[6];
  const float* ef_b2 = (const float*)d_in[7];
  const float* em_w1 = (const float*)d_in[8];
  const float* em_b1 = (const float*)d_in[9];
  const float* em_w2 = (const float*)d_in[10];
  const float* em_b2 = (const float*)d_in[11];

  float* out = (float*)d_out;
  float* emb = (float*)d_ws;  // 96 floats
  const long long n = (long long)in_sizes[2];  // N_POINTS from mask

  hipLaunchKernelGGL(setup_embed_kernel, dim3(1), dim3(32), 0, stream,
                     t, ef_w1, ef_b1, ef_w2, ef_b2, em_w1, em_b1, em_w2, em_b2, emb);

  const int nb = (int)((n + TPTS - 1) / TPTS);
  hipLaunchKernelGGL(scatter_kernel, dim3(nb), dim3(256), 0, stream,
                     pdf, mask, emb, out, n);
}
